// GeometricStructureEmbedding_33285996544414
// MI455X (gfx1250) — matrix operand order repaired and checked by reading, not compiled, not run on hardware
//
#include <hip/hip_runtime.h>
#include <hip/hip_bf16.h>
#include <math.h>

// GeometricStructureEmbedding for MI455X (gfx1250), wave32 + WMMA f16.
// out[i*1024+j, n] = sum_k emb(angle(i,j))[k] * W[n,k] + b[n]
// M=1M, N=K=128 GEMM with A generated on the fly from sin/cos.

typedef __attribute__((ext_vector_type(16))) _Float16 v16h;
typedef __attribute__((ext_vector_type(8)))  _Float16 v8h;
typedef __attribute__((ext_vector_type(4)))  _Float16 v4h;
typedef __attribute__((ext_vector_type(8)))  float    v8f;

union AU { v16h v; _Float16 e[16]; };
union BU { v16h v; v8h h[2]; };

#define HID   128
#define LPITCH 136   // LDS row pitch in halves: 272B = 17*16B -> conflict-friendly
#define FACTOR_A 3.81971863420549f           // 180/(15*pi)
#define NEG_LN1E4_OVER_64 (-0.14391156831f)  // -ln(10000)/64

__global__ __launch_bounds__(256) void geo_emb_wmma(
    const float* __restrict__ nrm,   // [Npts,3]
    const float* __restrict__ W,     // [128,128] row n, col k
    const float* __restrict__ bias,  // [128]
    float* __restrict__ out,         // [Npts*Npts,128]
    int Npts)
{
    __shared__ _Float16 Wlds[HID * LPITCH];  // ~34 KB of f16 W, pitched

    const int tid = threadIdx.x;

    // ---- Stage W (f32 -> f16) into LDS: 2 threads per row, 64 cols each ----
    {
        const int row  = tid >> 1;
        const int half = tid & 1;
        const float4* Wv = (const float4*)(W + row * HID + half * 64);
        _Float16* dst = &Wlds[row * LPITCH + half * 64];
        #pragma unroll
        for (int q = 0; q < 16; ++q) {
            float4 v = Wv[q];
            v4h hv = { (_Float16)v.x, (_Float16)v.y, (_Float16)v.z, (_Float16)v.w };
            *(v4h*)(dst + q * 4) = hv;
        }
    }
    __syncthreads();

    const int lane = tid & 31;
    const int w    = tid >> 5;       // wave id 0..7
    const int ncol = lane & 15;      // N (and j) column within tile
    const int hh   = lane >> 4;      // half-wave select

    const int i = blockIdx.x;        // reference row index

    // ---- Bias per N-tile (C/D layout: N = 16*nt + ncol for both halves) ----
    float bv[8];
    #pragma unroll
    for (int nt = 0; nt < 8; ++nt) bv[nt] = bias[nt * 16 + ncol];

    // ---- Reference normal (uniform within block) ----
    const float rx = nrm[i * 3 + 0];
    const float ry = nrm[i * 3 + 1];
    const float rz = nrm[i * 3 + 2];

    // ---- Hoisted div_term values for this lane's A-fragment slots ----
    // A 16-bit 16x32 layout: lane half hh covers K = 16g + 8*hh + {0..7} per k-step.
    // Pair (sin,cos) shares kp = K>>1.
    float dco[32];
    #pragma unroll
    for (int kk = 0; kk < 4; ++kk)
        #pragma unroll
        for (int g = 0; g < 2; ++g)
            #pragma unroll
            for (int q = 0; q < 4; ++q) {
                int kp = 16 * kk + 8 * g + 4 * hh + q;
                dco[kk * 8 + g * 4 + q] = __expf(NEG_LN1E4_OVER_64 * (float)kp);
            }

    const int tiles = Npts >> 7;     // j-tiles per wave (8 for Npts=1024)

    for (int t = 0; t < tiles; ++t) {
        const int jbase = (w * tiles + t) * 16;
        const int j = jbase + ncol;  // each lane owns pair M = ncol of the tile

        // ---- angle(i,j) = atan2(|n_i x n_j|, n_i . n_j) * FACTOR_A ----
        const float ax = nrm[j * 3 + 0];
        const float ay = nrm[j * 3 + 1];
        const float az = nrm[j * 3 + 2];
        const float cx = ry * az - rz * ay;
        const float cy = rz * ax - rx * az;
        const float cz = rx * ay - ry * ax;
        const float sv = sqrtf(cx * cx + cy * cy + cz * cz);
        const float cv = rx * ax + ry * ay + rz * az;
        const float ang = atan2f(sv, cv) * FACTOR_A;

        // ---- Build A fragments: emb[2kp]=sin(ang*d[kp]), emb[2kp+1]=cos ----
        AU a4[4];
        #pragma unroll
        for (int kk = 0; kk < 4; ++kk)
            #pragma unroll
            for (int g = 0; g < 2; ++g)
                #pragma unroll
                for (int q = 0; q < 4; ++q) {
                    float om = ang * dco[kk * 8 + g * 4 + q];
                    a4[kk].e[8 * g + 2 * q + 0] = (_Float16)__sinf(om);
                    a4[kk].e[8 * g + 2 * q + 1] = (_Float16)__cosf(om);
                }

        // ---- Accumulators seeded with bias ----
        v8f acc[8];
        #pragma unroll
        for (int nt = 0; nt < 8; ++nt)
            #pragma unroll
            for (int c = 0; c < 8; ++c) acc[nt][c] = bv[nt];

        // ---- 8 N-tiles x 4 K-steps of v_wmma_f32_16x16x32_f16 ----
        // B layout (32x16): element j of the frag -> K = 8*hh + j (j<8), 16 + 8*hh + (j-8) (j>=8),
        // N = ncol. B[k][n] = W[n][k].
        #pragma unroll
        for (int nt = 0; nt < 8; ++nt) {
            const _Float16* wr = &Wlds[(nt * 16 + ncol) * LPITCH + 8 * hh];
            #pragma unroll
            for (int kk = 0; kk < 4; ++kk) {
                BU bf;
                bf.h[0] = *(const v8h*)(wr + 32 * kk);
                bf.h[1] = *(const v8h*)(wr + 32 * kk + 16);
                acc[nt] = __builtin_amdgcn_wmma_f32_16x16x32_f16(
                    false, a4[kk].v, false, bf.v,
                    (short)0, acc[nt], false, false);
            }
        }

        // ---- Stream out 16x128 f32 slab (non-temporal: never re-read) ----
        // C layout: VGPR r -> row M = r (lanes 0-15) / r+8 (lanes 16-31), N = ncol.
        const size_t rowbase = (size_t)i * (size_t)Npts + (size_t)jbase;
        #pragma unroll
        for (int nt = 0; nt < 8; ++nt)
            #pragma unroll
            for (int r = 0; r < 8; ++r) {
                int row = r + 8 * hh;
                __builtin_nontemporal_store(
                    acc[nt][r],
                    &out[(rowbase + row) * HID + nt * 16 + ncol]);
            }
    }
}

extern "C" void kernel_launch(void* const* d_in, const int* in_sizes, int n_in,
                              void* d_out, int out_size, void* d_ws, size_t ws_size,
                              hipStream_t stream) {
    // inputs: [0]=points (unused), [1]=normals, [2]=W, [3]=b
    const float* normals = (const float*)d_in[1];
    const float* W       = (const float*)d_in[2];
    const float* b       = (const float*)d_in[3];
    float* out           = (float*)d_out;
    const int Npts = in_sizes[1] / 3;   // 1024

    dim3 grid(Npts), block(256);
    hipLaunchKernelGGL(geo_emb_wmma, grid, block, 0, stream,
                       normals, W, b, out, Npts);
}
